// TemporalAttention_37409165148452
// MI455X (gfx1250) — matrix operand order repaired and checked by reading, not compiled, not run on hardware
//
#include <hip/hip_runtime.h>

typedef __attribute__((ext_vector_type(16))) _Float16 v16h;
typedef __attribute__((ext_vector_type(8)))  float    v8f;
typedef __attribute__((ext_vector_type(4)))  float    v4f;
typedef __attribute__((ext_vector_type(4)))  unsigned int v4u;

union F16Frag { v16h h; v4u u[2]; };

#define TS 520   // token/q/k/att row stride (halfs) -> conflict-free b128 LDS reads
#define VS 32    // V-transposed row stride (halfs)
#define SCS 33   // score row stride (floats)
#define PS 40    // softmax-prob row stride (halfs), overlaid on score buffer

__device__ __forceinline__ v8f wmma_f16(v16h a, v16h b, v8f c) {
  return __builtin_amdgcn_wmma_f32_16x16x32_f16(false, a, false, b, (short)0, c, false, false);
}

// A fragment (16x32 f16): row-major source, row for this lane, K-base kb (mult of 32).
__device__ __forceinline__ v16h frag_a(const _Float16* row, int kb, int hs) {
  F16Frag f;
  f.u[0] = *(const v4u*)(row + kb + hs * 8);
  f.u[1] = *(const v4u*)(row + kb + 16 + hs * 8);
  return f.h;
}
// B fragment (32x16 f16): source is B^T row-major (row n = column n of B, K contiguous).
__device__ __forceinline__ v16h frag_b(const _Float16* rowT, int kb, int hs) {
  F16Frag f;
  f.u[0] = *(const v4u*)(rowT + kb + hs * 8);
  f.u[1] = *(const v4u*)(rowT + kb + 16 + hs * 8);
  return f.h;
}

// Prep: transpose + f16-convert weights into workspace; zero the mean-accumulator
// region of the output (ap_o / af_o) so the fused kernel can atomically accumulate.
__global__ __launch_bounds__(256) void ta_prep_kernel(
    const float* __restrict__ W_qkv, const float* __restrict__ W_proj,
    _Float16* __restrict__ WqT, _Float16* __restrict__ WpT, float* __restrict__ out) {
  const int idx = blockIdx.x * 256 + threadIdx.x;   // exactly 1,048,576 threads
  if (idx < 786432) {                               // WqT[n][k] = W_qkv[k][n], n<1536
    const int n = idx >> 9, k = idx & 511;
    WqT[idx] = (_Float16)W_qkv[k * 1536 + n];
  } else {                                          // WpT[n][k] = W_proj[k][n], n<512
    const int i2 = idx - 786432;
    const int n = i2 >> 9, k = i2 & 511;
    WpT[i2] = (_Float16)W_proj[k * 512 + n];
  }
  if (idx < 32768) out[8388608 + idx] = 0.0f;       // ap_o | af_o region
}

__global__ __launch_bounds__(256) void ta_fused_kernel(
    const float* __restrict__ v_p, const float* __restrict__ v_f,
    const float* __restrict__ a_p, const float* __restrict__ a_f,
    const _Float16* __restrict__ WqT, const _Float16* __restrict__ WpT,
    const float* __restrict__ b_proj, float* __restrict__ out) {
  __shared__ _Float16 sTok[32 * TS];    // tokens (f16); reused as attention output
  __shared__ _Float16 sQ[32 * TS];
  __shared__ _Float16 sK[32 * TS];
  __shared__ _Float16 sVT[512 * VS];    // V transposed: [feature][token]
  __shared__ float    sSC[4 * 32 * SCS];// scores for one 4-head group; P overlaid as f16

  const int tid  = threadIdx.x;
  const int lane = tid & 31;
  const int wave = tid >> 5;
  const int col  = lane & 15;
  const int hs   = (lane >> 4) & 1;
  const int b    = blockIdx.x >> 8;
  const int s    = blockIdx.x & 255;

  // ---- Phase 0: gather 32 tokens (v_p | v_f | a_p bcast | a_f bcast) -> LDS f16
  for (int idx = tid; idx < 4096; idx += 256) {
    const int t = idx >> 7;
    const int c = (idx & 127) << 2;
    const float* src;
    if (t < 8)       src = v_p + (size_t)((((b << 3) + t) << 8) + s) * 512 + c;
    else if (t < 16) src = v_f + (size_t)((((b << 3) + (t - 8)) << 8) + s) * 512 + c;
    else if (t < 24) src = a_p + (((b << 3) + (t - 16)) << 9) + c;
    else             src = a_f + (((b << 3) + (t - 24)) << 9) + c;
    v4f f = *(const v4f*)src;
    _Float16* dst = &sTok[t * TS + c];
    dst[0] = (_Float16)f.x; dst[1] = (_Float16)f.y;
    dst[2] = (_Float16)f.z; dst[3] = (_Float16)f.w;
  }
  __syncthreads();

  // ---- Phase 1: QKV GEMM  (32 x 512) @ (512 x 1536), W streamed from L2 as f16
  for (int nt = wave; nt < 96; nt += 8) {
    const int c = (nt << 4) + col;
    const _Float16* brow  = WqT + (size_t)c * 512;
    const _Float16* arow0 = &sTok[col * TS];
    const _Float16* arow1 = &sTok[(16 + col) * TS];
    v8f acc0 = {}; v8f acc1 = {};
#pragma unroll
    for (int kk = 0; kk < 16; ++kk) {
      const int kb = kk << 5;
      const v16h bf = frag_b(brow, kb, hs);
      acc0 = wmma_f16(frag_a(arow0, kb, hs), bf, acc0);
      acc1 = wmma_f16(frag_a(arow1, kb, hs), bf, acc1);
    }
#pragma unroll
    for (int r = 0; r < 8; ++r) {
      const int m0 = r + (hs << 3);
      const int m1 = m0 + 16;
      if (c < 512) {
        sQ[m0 * TS + c] = (_Float16)acc0[r];
        sQ[m1 * TS + c] = (_Float16)acc1[r];
      } else if (c < 1024) {
        sK[m0 * TS + (c - 512)] = (_Float16)acc0[r];
        sK[m1 * TS + (c - 512)] = (_Float16)acc1[r];
      } else {
        sVT[(c - 1024) * VS + m0] = (_Float16)acc0[r];
        sVT[(c - 1024) * VS + m1] = (_Float16)acc1[r];
      }
    }
  }
  __syncthreads();

  // ---- Phase 2: RoPE on q,k (half=32 rotate within each 64-wide head)
  for (int idx = tid; idx < 8192; idx += 256) {
    const int i = idx & 31;
    const int h = (idx >> 5) & 7;
    const int t = idx >> 8;
    const int c1 = (h << 6) + i, c2 = c1 + 32;
    const float ang = (float)t * __expf(-0.2878231366242557f * (float)i); // 10000^(-i/32)
    float sn, cs; __sincosf(ang, &sn, &cs);
    const float q1 = (float)sQ[t * TS + c1], q2 = (float)sQ[t * TS + c2];
    sQ[t * TS + c1] = (_Float16)(q1 * cs - q2 * sn);
    sQ[t * TS + c2] = (_Float16)(q2 * cs + q1 * sn);
    const float k1 = (float)sK[t * TS + c1], k2 = (float)sK[t * TS + c2];
    sK[t * TS + c1] = (_Float16)(k1 * cs - k2 * sn);
    sK[t * TS + c2] = (_Float16)(k2 * cs + k1 * sn);
  }
  __syncthreads();

  // ---- Phase 3: attention in two groups of 4 heads (keeps LDS small)
  for (int hg = 0; hg < 2; ++hg) {
    // scores = q @ k^T (per head 32x32, K=64): 4 heads x 4 tiles
    for (int tile = wave; tile < 16; tile += 8) {
      const int hl = tile >> 2;
      const int h  = (hg << 2) + hl;
      const int m0 = ((tile >> 1) & 1) << 4;
      const int n0 = (tile & 1) << 4;
      const int kbase = h << 6;
      const _Float16* arow = &sQ[(m0 + col) * TS + kbase];
      const _Float16* brow = &sK[(n0 + col) * TS + kbase];
      v8f acc = {};
      acc = wmma_f16(frag_a(arow, 0, hs),  frag_b(brow, 0, hs),  acc);
      acc = wmma_f16(frag_a(arow, 32, hs), frag_b(brow, 32, hs), acc);
#pragma unroll
      for (int r = 0; r < 8; ++r)
        sSC[((hl << 5) + m0 + r + (hs << 3)) * SCS + n0 + col] = acc[r];
    }
    __syncthreads();

    // softmax: one thread per (head-in-group, row); scale = 64^-0.5 = 0.125
    float rowv[32]; float inv = 0.f;
    const bool act = tid < 128;
    const int hl = tid >> 5, t = tid & 31;
    if (act) {
      const float* srow = &sSC[((hl << 5) + t) * SCS];
      float mx = -3.0e38f;
#pragma unroll
      for (int j = 0; j < 32; ++j) { rowv[j] = srow[j] * 0.125f; mx = fmaxf(mx, rowv[j]); }
      float sum = 0.f;
#pragma unroll
      for (int j = 0; j < 32; ++j) { rowv[j] = __expf(rowv[j] - mx); sum += rowv[j]; }
      inv = 1.0f / sum;
    }
    __syncthreads();  // all score reads done before overlaying P (f16) on sSC
    if (act) {
      _Float16* prow = ((_Float16*)sSC) + ((hl << 5) + t) * PS;
#pragma unroll
      for (int j = 0; j < 32; ++j) prow[j] = (_Float16)(rowv[j] * inv);
    }
    __syncthreads();

    // out = P @ V (per head 32x64, K=32): 4 heads x 8 tiles; result into sTok (reuse)
    const _Float16* sP = (const _Float16*)sSC;
    for (int tile = wave; tile < 32; tile += 8) {
      const int hl2 = tile >> 3;
      const int h   = (hg << 2) + hl2;
      const int m0  = ((tile >> 2) & 1) << 4;
      const int f0  = (h << 6) + ((tile & 3) << 4);
      const _Float16* arow = sP + ((hl2 << 5) + m0 + col) * PS;
      const _Float16* brow = &sVT[(f0 + col) * VS];
      v8f acc = {};
      acc = wmma_f16(frag_a(arow, 0, hs), frag_b(brow, 0, hs), acc);
#pragma unroll
      for (int r = 0; r < 8; ++r)
        sTok[(m0 + r + (hs << 3)) * TS + f0 + col] = (_Float16)acc[r];
    }
    __syncthreads();
  }

  // ---- Phase 4: proj GEMM (32 x 512) @ (512 x 512) + bias, scatter outputs
  constexpr size_t VF_OFF = 4194304, AP_OFF = 8388608, AF_OFF = 8404992;
  for (int nt = wave; nt < 32; nt += 8) {
    const int c = (nt << 4) + col;
    const _Float16* brow  = WpT + (size_t)c * 512;
    const _Float16* arow0 = &sTok[col * TS];
    const _Float16* arow1 = &sTok[(16 + col) * TS];
    v8f acc0 = {}; v8f acc1 = {};
#pragma unroll
    for (int kk = 0; kk < 16; ++kk) {
      const int kb = kk << 5;
      const v16h bf = frag_b(brow, kb, hs);
      acc0 = wmma_f16(frag_a(arow0, kb, hs), bf, acc0);
      acc1 = wmma_f16(frag_a(arow1, kb, hs), bf, acc1);
    }
    const float bias = b_proj[c];
#pragma unroll
    for (int r = 0; r < 8; ++r) {
      const int t0 = r + (hs << 3);          // tokens 0..15 -> vp_o / vf_o
      const int t1 = t0 + 16;                // tokens 16..31 -> ap_o / af_o means
      const float o0 = acc0[r] + bias;
      const float o1 = acc1[r] + bias;
      if (t0 < 8)
        out[(size_t)((((b << 3) + t0) << 8) + s) * 512 + c] = o0;
      else
        out[VF_OFF + (size_t)((((b << 3) + (t0 - 8)) << 8) + s) * 512 + c] = o0;
      if (t1 < 24)
        atomicAdd(&out[AP_OFF + (((b << 3) + (t1 - 16)) << 9) + c], o1 * 0.00390625f);
      else
        atomicAdd(&out[AF_OFF + (((b << 3) + (t1 - 24)) << 9) + c], o1 * 0.00390625f);
    }
  }
}

extern "C" void kernel_launch(void* const* d_in, const int* in_sizes, int n_in,
                              void* d_out, int out_size, void* d_ws, size_t ws_size,
                              hipStream_t stream) {
  const float* v_p    = (const float*)d_in[0];
  const float* v_f    = (const float*)d_in[1];
  const float* a_p    = (const float*)d_in[2];
  const float* a_f    = (const float*)d_in[3];
  const float* W_qkv  = (const float*)d_in[4];
  const float* W_proj = (const float*)d_in[5];
  const float* b_proj = (const float*)d_in[6];
  float* out = (float*)d_out;
  _Float16* WqT = (_Float16*)d_ws;          // 1536*512 f16
  _Float16* WpT = WqT + 786432;             // 512*512 f16  (total ws use: 2 MiB)

  ta_prep_kernel<<<4096, 256, 0, stream>>>(W_qkv, W_proj, WqT, WpT, out);
  ta_fused_kernel<<<1024, 256, 0, stream>>>(v_p, v_f, a_p, a_f, WqT, WpT, b_proj, out);
  (void)in_sizes; (void)n_in; (void)out_size; (void)ws_size;
}
